// MultiNodeClassificationGroupHead_64312840290551
// MI455X (gfx1250) — matrix operand order repaired and checked by reading, not compiled, not run on hardware
//
#include <hip/hip_runtime.h>

#define D      256
#define A_FEAT 16
#define KP     288      // K for layer 1 padded: 272 -> 288 (9 x 32)
#define C_OUT  10
#define TM     128      // rows per block
#define NTHR   256      // 8 waves of 32
#define WTILE  (D * 32) // one staged weight K-tile: 256 rows x 32 bf16

typedef __bf16 bf16;
typedef __attribute__((ext_vector_type(16))) __bf16        v16bf;
typedef __attribute__((ext_vector_type(8)))  float         v8f;
typedef __attribute__((ext_vector_type(4)))  unsigned int  u32x4;

union Frag { u32x4 u[2]; v16bf v; };

__device__ __forceinline__ float gelu_exact(float x) {
    return 0.5f * x * (1.0f + erff(x * 0.70710678118654752f));
}

// A fragment (16x32 bf16): lanes 0-15 -> M=lane, chunks K=[k0..k0+7],[k0+16..k0+23]
//                          lanes16-31 -> M=lane-16, chunks K=[k0+8..+15],[k0+24..+31]
__device__ __forceinline__ v16bf load_frag_a(const bf16* base, int stride, int row,
                                             int k0, int lane) {
    const int c0 = (lane >> 4) << 3;            // 0 or 8
    const bf16* p = base + row * stride + k0 + c0;
    Frag f;
    f.u[0] = *(const u32x4*)(p);
    f.u[1] = *(const u32x4*)(p + 16);
    return f.v;
}

// B fragment (32x16 bf16) from W^T[N][K] rows: lanes 0-15 -> N=lane, K=[k0..k0+7],[k0+16..k0+23]
//                                              lanes16-31 -> N=lane-16, K=[k0+8..+15],[k0+24..+31]
__device__ __forceinline__ v16bf load_frag_b(const bf16* base, int stride, int nrow,
                                             int k0, int lane) {
    const int kc = (lane >> 4) << 3;            // 0 or 8
    const bf16* p = base + nrow * stride + k0 + kc;
    Frag f;
    f.u[0] = *(const u32x4*)(p);
    f.u[1] = *(const u32x4*)(p + 16);
    return f.v;
}

__device__ __forceinline__ v8f wmma_bf16(v16bf a, v16bf b, v8f c) {
    return __builtin_amdgcn_wmma_f32_16x16x32_bf16(false, a, false, b,
                                                   (short)0, c, false, false);
}

// --- CDNA5 async global->LDS copy (ASYNCcnt-tracked), inline asm for
//     toolchain portability. In GV mode the instruction immediate offset is
//     added to BOTH the LDS and global address, so one base pair covers the
//     whole 64-byte row.
__device__ __forceinline__ uint32_t lds_off32(const void* p) {
    return (uint32_t)(uintptr_t)p;   // low 32 bits of flat addr == LDS offset
}

__device__ __forceinline__ void async_copy_row64B(uint32_t lds_off, const bf16* gsrc) {
    asm volatile(
        "global_load_async_to_lds_b128 %0, %1, off\n\t"
        "global_load_async_to_lds_b128 %0, %1, off offset:16\n\t"
        "global_load_async_to_lds_b128 %0, %1, off offset:32\n\t"
        "global_load_async_to_lds_b128 %0, %1, off offset:48"
        :: "v"(lds_off), "v"(gsrc) : "memory");
}

__device__ __forceinline__ void wait_async0() {
    asm volatile("s_wait_asynccnt 0x0" ::: "memory");
}

// ---------------------------------------------------------------------------
// prep: zero sums/counts, build bf16 transposed (and padded) weights in ws
// ---------------------------------------------------------------------------
__global__ void prep_kernel(const float* __restrict__ W1a, const float* __restrict__ W2a,
                            const float* __restrict__ Wc1, const float* __restrict__ Wc2,
                            float* __restrict__ sums, float* __restrict__ counts,
                            bf16* __restrict__ w1aT, bf16* __restrict__ w2aT,
                            bf16* __restrict__ wc1T, bf16* __restrict__ wc2T, int G) {
    const int tid = blockIdx.x * blockDim.x + threadIdx.x;
    const int stride = gridDim.x * blockDim.x;

    const int zn = G * D + G;
    for (int i = tid; i < zn; i += stride) {
        if (i < G * D) sums[i] = 0.0f; else counts[i - G * D] = 0.0f;
    }
    for (int i = tid; i < D * KP; i += stride) {            // W1a [272,256] -> [256,288]
        int n = i / KP, k = i - n * KP;
        float v = (k < (D + A_FEAT)) ? W1a[k * D + n] : 0.0f;
        w1aT[n * KP + k] = (bf16)v;
    }
    for (int i = tid; i < D * D; i += stride) {             // W2a, Wc1 [256,256] -> T
        int n = i / D, k = i - n * D;
        w2aT[n * D + k] = (bf16)W2a[k * D + n];
        wc1T[n * D + k] = (bf16)Wc1[k * D + n];
    }
    for (int i = tid; i < 16 * D; i += stride) {            // Wc2 [256,10] -> [16,256]
        int n = i / D, k = i - n * D;
        float v = (n < C_OUT) ? Wc2[k * C_OUT + n] : 0.0f;
        wc2T[n * D + k] = (bf16)v;
    }
}

// ---------------------------------------------------------------------------
// node MLP (two bf16 WMMA GEMMs) + segment-sum accumulation via f32 atomics
// ---------------------------------------------------------------------------
__global__ void __launch_bounds__(NTHR)
mlp_kernel(const float* __restrict__ h, const float* __restrict__ sf,
           const bf16* __restrict__ w1aT, const bf16* __restrict__ w2aT,
           const float* __restrict__ b1a, const float* __restrict__ b2a,
           const int* __restrict__ seg,
           float* __restrict__ sums, float* __restrict__ counts, int N) {
    extern __shared__ char smem[];
    bf16* lds_x  = (bf16*)smem;                               // [TM][KP] -> reused as Y1 [TM][D]
    bf16* lds_w0 = (bf16*)(smem + TM * KP * 2);               // W tile ping
    bf16* lds_w1 = (bf16*)(smem + TM * KP * 2 + WTILE * 2);   // W tile pong

    const int tid  = threadIdx.x;
    const int lane = tid & 31;
    const int wave = tid >> 5;
    const int rowBase = blockIdx.x * TM;
    const v8f vzero = {0.f, 0.f, 0.f, 0.f, 0.f, 0.f, 0.f, 0.f};
    const uint32_t woff0 = lds_off32(lds_w0 + tid * 32);
    const uint32_t woff1 = lds_off32(lds_w1 + tid * 32);

    // prologue: async-stage W1a K-tile 0 while we convert the X tile
    async_copy_row64B(woff0, w1aT + tid * KP);

    // ---- load X tile as bf16: [0,256)=h, [256,272)=stage_feats, [272,288)=0
    for (int idx = tid; idx < TM * KP; idx += NTHR) {
        int r = idx / KP, c = idx - r * KP;
        int row = rowBase + r;
        float v = 0.0f;
        if (row < N) {
            if (c < D)               v = h[(size_t)row * D + c];
            else if (c < D + A_FEAT) v = sf[(size_t)row * A_FEAT + (c - D)];
        }
        lds_x[idx] = (bf16)v;
    }

    v8f acc[16];
    #pragma unroll
    for (int t = 0; t < 16; ++t) acc[t] = vzero;

    // ---- GEMM 1: Y1 = X[128,288] @ W1a[288,256]   (double-buffered W tiles)
    const int steps1 = KP / 32;
    for (int ks = 0; ks < steps1; ++ks) {
        wait_async0();
        __syncthreads();
        if (ks + 1 < steps1)
            async_copy_row64B((ks & 1) ? woff0 : woff1,
                              w1aT + tid * KP + (ks + 1) * 32);
        const bf16* wb = (ks & 1) ? lds_w1 : lds_w0;
        v16bf a = load_frag_a(lds_x, KP, wave * 16 + (lane & 15), ks * 32, lane);
        #pragma unroll
        for (int nt = 0; nt < 16; ++nt) {
            v16bf b = load_frag_b(wb, 32, nt * 16 + (lane & 15), 0, lane);
            acc[nt] = wmma_bf16(a, b, acc[nt]);
        }
    }

    // ---- bias + exact GELU, write Y1 (bf16) into the X buffer
    __syncthreads();                                  // all waves done with lds_x / W tiles
    async_copy_row64B(woff0, w2aT + tid * D);         // pre-stage W2a K-tile 0
    #pragma unroll
    for (int nt = 0; nt < 16; ++nt) {
        const int n = nt * 16 + (lane & 15);
        const float bias = b1a[n];
        #pragma unroll
        for (int v = 0; v < 8; ++v) {
            const int m = wave * 16 + v + ((lane >> 4) << 3);
            lds_x[m * D + n] = (bf16)gelu_exact(acc[nt][v] + bias);
        }
        acc[nt] = vzero;
    }

    // ---- GEMM 2: X2 = Y1[128,256] @ W2a[256,256]
    const int steps2 = D / 32;
    for (int ks = 0; ks < steps2; ++ks) {
        wait_async0();
        __syncthreads();
        if (ks + 1 < steps2)
            async_copy_row64B((ks & 1) ? woff0 : woff1,
                              w2aT + tid * D + (ks + 1) * 32);
        const bf16* wb = (ks & 1) ? lds_w1 : lds_w0;
        v16bf a = load_frag_a(lds_x, D, wave * 16 + (lane & 15), ks * 32, lane);
        #pragma unroll
        for (int nt = 0; nt < 16; ++nt) {
            v16bf b = load_frag_b(wb, 32, nt * 16 + (lane & 15), 0, lane);
            acc[nt] = wmma_bf16(a, b, acc[nt]);
        }
    }

    // ---- bias + segment-sum accumulation (ids sorted -> hot L2 lines)
    #pragma unroll
    for (int nt = 0; nt < 16; ++nt) {
        const int n = nt * 16 + (lane & 15);
        const float bias = b2a[n];
        #pragma unroll
        for (int v = 0; v < 8; ++v) {
            const int m = wave * 16 + v + ((lane >> 4) << 3);
            const int row = rowBase + m;
            if (row < N) {
                const int s = seg[row];
                atomicAdd(&sums[(size_t)s * D + n], acc[nt][v] + bias);
            }
        }
    }
    if (tid < TM) {
        const int row = rowBase + tid;
        if (row < N) atomicAdd(&counts[seg[row]], 1.0f);
    }
}

// ---------------------------------------------------------------------------
// classifier head: mean -> GELU(grouped@Wc1+bc1) @ Wc2 + bc2
// ---------------------------------------------------------------------------
__global__ void __launch_bounds__(NTHR)
head_kernel(const float* __restrict__ sums, const float* __restrict__ counts,
            const bf16* __restrict__ wc1T, const bf16* __restrict__ wc2T,
            const float* __restrict__ bc1, const float* __restrict__ bc2,
            float* __restrict__ out, int G) {
    extern __shared__ char smem[];
    bf16* lds_x  = (bf16*)smem;                               // [TM][D], reused for T
    bf16* lds_w0 = (bf16*)(smem + TM * D * 2);
    bf16* lds_w1 = (bf16*)(smem + TM * D * 2 + WTILE * 2);

    const int tid  = threadIdx.x;
    const int lane = tid & 31;
    const int wave = tid >> 5;
    const int gBase = blockIdx.x * TM;
    const v8f vzero = {0.f, 0.f, 0.f, 0.f, 0.f, 0.f, 0.f, 0.f};
    const uint32_t woff0 = lds_off32(lds_w0 + tid * 32);
    const uint32_t woff1 = lds_off32(lds_w1 + tid * 32);

    async_copy_row64B(woff0, wc1T + tid * D);     // pre-stage Wc1 K-tile 0

    // ---- grouped mean tile (bf16)
    for (int idx = tid; idx < TM * D; idx += NTHR) {
        int r = idx >> 8, c = idx & (D - 1);
        int g = gBase + r;
        float v = 0.0f;
        if (g < G) v = sums[(size_t)g * D + c] / fmaxf(counts[g], 1.0f);
        lds_x[idx] = (bf16)v;
    }

    v8f acc[16];
    #pragma unroll
    for (int t = 0; t < 16; ++t) acc[t] = vzero;

    // ---- GEMM: T = grouped[128,256] @ Wc1[256,256]
    const int steps = D / 32;
    for (int ks = 0; ks < steps; ++ks) {
        wait_async0();
        __syncthreads();
        if (ks + 1 < steps)
            async_copy_row64B((ks & 1) ? woff0 : woff1,
                              wc1T + tid * D + (ks + 1) * 32);
        const bf16* wb = (ks & 1) ? lds_w1 : lds_w0;
        v16bf a = load_frag_a(lds_x, D, wave * 16 + (lane & 15), ks * 32, lane);
        #pragma unroll
        for (int nt = 0; nt < 16; ++nt) {
            v16bf b = load_frag_b(wb, 32, nt * 16 + (lane & 15), 0, lane);
            acc[nt] = wmma_bf16(a, b, acc[nt]);
        }
    }

    __syncthreads();
    #pragma unroll
    for (int nt = 0; nt < 16; ++nt) {
        const int n = nt * 16 + (lane & 15);
        const float bias = bc1[n];
        #pragma unroll
        for (int v = 0; v < 8; ++v) {
            const int m = wave * 16 + v + ((lane >> 4) << 3);
            lds_x[m * D + n] = (bf16)gelu_exact(acc[nt][v] + bias);
        }
    }
    __syncthreads();

    // ---- final: T[128,256] @ Wc2[256,16] (one column tile per wave, B from L2)
    v8f o = vzero;
    for (int ks = 0; ks < steps; ++ks) {
        const int k0 = ks * 32;
        v16bf a = load_frag_a(lds_x, D, wave * 16 + (lane & 15), k0, lane);
        v16bf b = load_frag_b(wc2T, D, lane & 15, k0, lane);
        o = wmma_bf16(a, b, o);
    }
    #pragma unroll
    for (int v = 0; v < 8; ++v) {
        const int m = wave * 16 + v + ((lane >> 4) << 3);
        const int n = lane & 15;
        const int g = gBase + m;
        if (g < G && n < C_OUT) out[(size_t)g * C_OUT + n] = o[v] + bc2[n];
    }
}

// ---------------------------------------------------------------------------
extern "C" void kernel_launch(void* const* d_in, const int* in_sizes, int n_in,
                              void* d_out, int out_size, void* d_ws, size_t ws_size,
                              hipStream_t stream) {
    const float* h   = (const float*)d_in[0];
    const float* sf  = (const float*)d_in[1];
    const float* W1a = (const float*)d_in[2];
    const float* b1a = (const float*)d_in[3];
    const float* W2a = (const float*)d_in[4];
    const float* b2a = (const float*)d_in[5];
    const float* Wc1 = (const float*)d_in[6];
    const float* bc1 = (const float*)d_in[7];
    const float* Wc2 = (const float*)d_in[8];
    const float* bc2 = (const float*)d_in[9];
    const int*   seg = (const int*)d_in[10];

    const int N = in_sizes[0] / D;           // 500000
    const int G = out_size / C_OUT;          // 100000
    float* out = (float*)d_out;

    // workspace carve-up
    char* ws = (char*)d_ws;
    float* sums   = (float*)ws;                                   // G*D f32
    float* counts = (float*)(ws + (size_t)G * D * 4);             // G f32
    size_t offW = (size_t)G * D * 4 + (((size_t)G * 4 + 255) & ~(size_t)255);
    bf16* w1aT = (bf16*)(ws + offW);                              // [256][288]
    bf16* w2aT = w1aT + D * KP;                                   // [256][256]
    bf16* wc1T = w2aT + D * D;                                    // [256][256]
    bf16* wc2T = wc1T + D * D;                                    // [16][256]
    (void)ws_size; (void)n_in;

    prep_kernel<<<4096, 256, 0, stream>>>(W1a, W2a, Wc1, Wc2, sums, counts,
                                          w1aT, w2aT, wc1T, wc2T, G);

    const int nb1 = (N + TM - 1) / TM;
    const size_t shm1 = (size_t)TM * KP * 2 + 2 * (size_t)WTILE * 2; // 106496 B
    mlp_kernel<<<nb1, NTHR, shm1, stream>>>(h, sf, w1aT, w2aT, b1a, b2a,
                                            seg, sums, counts, N);

    const int nb2 = (G + TM - 1) / TM;
    const size_t shm2 = (size_t)TM * D * 2 + 2 * (size_t)WTILE * 2;  // 98304 B
    head_kernel<<<nb2, NTHR, shm2, stream>>>(sums, counts, wc1T, wc2T,
                                             bc1, bc2, out, G);
}
